// GCN_with_feature_79147657330992
// MI455X (gfx1250) — compile-verified, matrix operand order checked by reading, not run on hardware
//
#include <hip/hip_runtime.h>

// ---------------------------------------------------------------------------
// GCN (3-step propagation) for MI455X / gfx1250, wave32, WMMA bf16.
//  - activations & weights pre-packed into WMMA register layouts (bf16, RNE)
//  - GEMM: 2 row-tiles per block, B reused across both; fused epilogue writes
//    both xw (for edge gather) and out = bias + dis^2*xw (+ res*ori)
//  - edge aggregation: L2-resident f32 hardware atomics
// ---------------------------------------------------------------------------

typedef __attribute__((ext_vector_type(16))) __bf16 v16bf;
typedef __attribute__((ext_vector_type(8)))  float  v8f;

#define HFEAT 128
#define TB 256

union ABRegs {
    unsigned u[8];
    uint4    q[2];
    v16bf    v;
};

// round-to-nearest-even f32 -> bf16 pair packed into one dword (lo in [15:0])
__device__ __forceinline__ unsigned pack_bf16(float lo, float hi) {
    unsigned ul = __float_as_uint(lo);
    unsigned uh = __float_as_uint(hi);
    ul += 0x7FFFu + ((ul >> 16) & 1u);
    uh += 0x7FFFu + ((uh >> 16) & 1u);
    return (ul >> 16) | (uh & 0xFFFF0000u);
}

// K index held by (vgpr j, lane-half h) for 16-bit A/B WMMA operands (K-block of 32)
__device__ __forceinline__ int kmap(int j, int h) {
    return (j < 4) ? (2 * j + 8 * h) : (16 + 2 * (j - 4) + 8 * h);
}

// --------------------------- degree / norm ---------------------------------

__global__ void k_deg_init(float* __restrict__ deg, int n) {
    int i = blockIdx.x * TB + threadIdx.x;
    if (i < n) deg[i] = 1.0f;              // self-loop weight
}

__global__ void k_deg_acc(const int* __restrict__ dst, const float* __restrict__ w,
                          float* __restrict__ deg, int ne) {
    int e = blockIdx.x * TB + threadIdx.x;
    if (e < ne) unsafeAtomicAdd(&deg[dst[e]], w[e]);
}

__global__ void k_dis(float* __restrict__ deg, int n) {
    int i = blockIdx.x * TB + threadIdx.x;
    if (i < n) {
        float d = deg[i];
        deg[i] = (d > 0.0f) ? rsqrtf(d) : 0.0f;   // in-place deg -> dis
    }
}

__global__ void k_norm(const int* __restrict__ src, const int* __restrict__ dst,
                       const float* __restrict__ w, const float* __restrict__ dis,
                       float* __restrict__ nrm, int ne) {
    int e = blockIdx.x * TB + threadIdx.x;
    if (e < ne) nrm[e] = dis[src[e]] * w[e] * dis[dst[e]];
}

// ------------------- weight pre-pack: f32 row-major -> WMMA B regs ----------
// Layout: Bp[tile_n(8)][kblock(4)][lane(32)][j(8)]  (8192 dwords = 32 KB)

__global__ void k_wconv(const float* __restrict__ W, unsigned* __restrict__ Bp) {
    int idx = blockIdx.x * TB + threadIdx.x;
    if (idx >= 8 * 4 * 32 * 8) return;
    int j    = idx & 7;
    int lane = (idx >> 3) & 31;
    int kb   = (idx >> 8) & 3;
    int t    = idx >> 10;
    int n  = t * 16 + (lane & 15);
    int h  = lane >> 4;
    int k0 = kb * 32 + kmap(j, h);
    Bp[idx] = pack_bf16(W[k0 * HFEAT + n], W[(k0 + 1) * HFEAT + n]);
}

// ----------- activation pre-pack: f32 [N,128] -> WMMA A regs (bf16) --------
// Layout: Ap[row_tile][kblock(4)][lane(32)][j(8)]; ReLU fused.

__global__ void k_prep(const float* __restrict__ X, unsigned* __restrict__ Ap,
                       int n_rows, int n_tiles, int do_relu) {
    int idx = blockIdx.x * TB + threadIdx.x;
    if (idx >= n_tiles * 1024) return;     // 4*32*8 dwords per 16-row tile
    int j    = idx & 7;
    int lane = (idx >> 3) & 31;
    int kb   = (idx >> 8) & 3;
    int t    = idx >> 10;
    int m = lane & 15, h = lane >> 4;
    int row = t * 16 + m;
    float lo = 0.0f, hi = 0.0f;
    if (row < n_rows) {
        int k = kb * 32 + kmap(j, h);
        lo = X[row * HFEAT + k];
        hi = X[row * HFEAT + k + 1];
        if (do_relu) { lo = fmaxf(lo, 0.0f); hi = fmaxf(hi, 0.0f); }
    }
    Ap[idx] = pack_bf16(lo, hi);
}

// ----------------------------- WMMA GEMM -----------------------------------
// Block = 2 row-tiles (32 rows) x 128 cols, 8 waves; each wave: per K-block
// load B once (2x b128) + A for both tiles (4x b128), 2 WMMAs. Epilogue
// stores raw xw and fused out = bias + dis^2*xw (+ res*ori).

template <bool HAS_ORI>
__global__ void __launch_bounds__(TB)
k_gemm(const unsigned* __restrict__ Ap, const unsigned* __restrict__ Bp,
       const float* __restrict__ dis, const float* __restrict__ bias,
       const float* __restrict__ ori, float res_scale,
       float* __restrict__ xw, float* __restrict__ out, int n_rows) {
    int t0   = blockIdx.x * 2;             // first of two row tiles
    int wave = threadIdx.x >> 5;           // column tile 0..7
    int lane = threadIdx.x & 31;
    int m = lane & 15;
    int h = lane >> 4;

    const unsigned* ap0 = Ap + (((size_t)t0 * 128 + lane) << 3);  // tile stride 1024 dw
    const unsigned* ap1 = ap0 + 1024;
    const unsigned* bp  = Bp + (((unsigned)wave * 128 + lane) << 3);

    v8f c0 = {}, c1 = {};
#pragma unroll
    for (int kb = 0; kb < 4; ++kb) {
        ABRegs a0, a1, b;
        b.q[0]  = *(const uint4*)(bp  + kb * 256);
        b.q[1]  = *(const uint4*)(bp  + kb * 256 + 4);
        a0.q[0] = *(const uint4*)(ap0 + kb * 256);
        a0.q[1] = *(const uint4*)(ap0 + kb * 256 + 4);
        a1.q[0] = *(const uint4*)(ap1 + kb * 256);
        a1.q[1] = *(const uint4*)(ap1 + kb * 256 + 4);
        c0 = __builtin_amdgcn_wmma_f32_16x16x32_bf16(false, a0.v, false, b.v,
                                                     (short)0, c0, false, false);
        c1 = __builtin_amdgcn_wmma_f32_16x16x32_bf16(false, a1.v, false, b.v,
                                                     (short)0, c1, false, false);
    }

    int col = wave * 16 + m;
    float bc = bias[col];

#pragma unroll
    for (int tt = 0; tt < 2; ++tt) {
        int t = t0 + tt;
        const v8f& c = tt ? c1 : c0;
        int rbase = t * 16 + 8 * h;        // vgpr r -> row rbase + r
        if ((t + 1) * 16 <= n_rows) {      // uniform: full tile, unguarded stores
#pragma unroll
            for (int r = 0; r < 8; ++r) {
                int row = rbase + r;
                float d = dis[row];
                float v = fmaf(d * d, c[r], bc);
                if (HAS_ORI) v = fmaf(res_scale, ori[row * HFEAT + col], v);
                xw[row * HFEAT + col]  = c[r];
                out[row * HFEAT + col] = v;
            }
        } else {
#pragma unroll
            for (int r = 0; r < 8; ++r) {
                int row = rbase + r;
                if (row < n_rows) {
                    float d = dis[row];
                    float v = fmaf(d * d, c[r], bc);
                    if (HAS_ORI) v = fmaf(res_scale, ori[row * HFEAT + col], v);
                    xw[row * HFEAT + col]  = c[r];
                    out[row * HFEAT + col] = v;
                }
            }
        }
    }
}

// --------------------------- edge aggregation ------------------------------
// one wave per edge; lane handles 4 consecutive features (float4, L2-resident)

__global__ void k_scatter(const int* __restrict__ src, const int* __restrict__ dst,
                          const float* __restrict__ nrm, const float* __restrict__ xw,
                          float* __restrict__ out, int n_edges) {
    int gtid = blockIdx.x * TB + threadIdx.x;
    int e = gtid >> 5;
    if (e >= n_edges) return;
    int lane = gtid & 31;
    float w = nrm[e];
    int s = src[e], d = dst[e];
    float4 v = ((const float4*)(xw + s * HFEAT))[lane];
    float* o = out + d * HFEAT + lane * 4;
    unsafeAtomicAdd(o + 0, w * v.x);
    unsafeAtomicAdd(o + 1, w * v.y);
    unsafeAtomicAdd(o + 2, w * v.z);
    unsafeAtomicAdd(o + 3, w * v.w);
}

// ----------------------------- launcher ------------------------------------

extern "C" void kernel_launch(void* const* d_in, const int* in_sizes, int n_in,
                              void* d_out, int out_size, void* d_ws, size_t ws_size,
                              hipStream_t stream) {
    const int*   eidx    = (const int*)d_in[0];
    const int    Ee      = in_sizes[0] / 2;
    const int*   src     = eidx;
    const int*   dst     = eidx + Ee;
    const float* in_feat = (const float*)d_in[1];
    const int    Nn      = in_sizes[1] / HFEAT;
    const float* ew      = (const float*)d_in[2];
    const float* W1      = (const float*)d_in[3];
    const float* b1      = (const float*)d_in[4];
    const float* W2      = (const float*)d_in[5];
    const float* b2      = (const float*)d_in[6];
    float*       out     = (float*)d_out;
    (void)n_in; (void)out_size; (void)ws_size;

    const int n_tiles    = (Nn + 15) / 16;
    const int n_tiles_ev = (n_tiles + 1) & ~1;     // pad so 2-tile blocks stay in carve

    // workspace carve-out (256B aligned)
    char*  ws  = (char*)d_ws;
    size_t off = 0;
    auto carve = [&](size_t bytes) {
        size_t p = off;
        off += (bytes + 255) & ~(size_t)255;
        return p;
    };
    float*    deg = (float*)(ws + carve((size_t)Nn * 4));            // becomes dis
    float*    nrm = (float*)(ws + carve((size_t)Ee * 4));
    float*    xw  = (float*)(ws + carve((size_t)Nn * HFEAT * 4));
    float*    ori = (float*)(ws + carve((size_t)Nn * HFEAT * 4));
    unsigned* Apk = (unsigned*)(ws + carve((size_t)n_tiles_ev * 1024 * 4));
    unsigned* Wb1 = (unsigned*)(ws + carve(8192 * 4));
    unsigned* Wb2 = (unsigned*)(ws + carve(8192 * 4));

    const int node_blk = (Nn + TB - 1) / TB;
    const int edge_blk = (Ee + TB - 1) / TB;
    const int prep_blk = (n_tiles * 1024 + TB - 1) / TB;
    const int gemm_blk = n_tiles_ev / 2;
    const int scat_blk = (int)(((long)Ee * 32 + TB - 1) / TB);

    // normalization: deg = 1 + sum_in w ; dis = rsqrt(deg) ; norm per edge
    k_deg_init<<<node_blk, TB, 0, stream>>>(deg, Nn);
    k_deg_acc <<<edge_blk, TB, 0, stream>>>(dst, ew, deg, Ee);
    k_dis     <<<node_blk, TB, 0, stream>>>(deg, Nn);
    k_norm    <<<edge_blk, TB, 0, stream>>>(src, dst, ew, deg, nrm, Ee);

    // pre-pack weights into WMMA B-register layout (bf16)
    k_wconv<<<32, TB, 0, stream>>>(W1, Wb1);
    k_wconv<<<32, TB, 0, stream>>>(W2, Wb2);

    // conv1: h1 = agg(in_feat @ W1) + b1      -> ori
    k_prep        <<<prep_blk, TB, 0, stream>>>(in_feat, Apk, Nn, n_tiles, 0);
    k_gemm<false> <<<gemm_blk, TB, 0, stream>>>(Apk, Wb1, deg, b1, nullptr, 0.0f,
                                                xw, ori, Nn);
    k_scatter     <<<scat_blk, TB, 0, stream>>>(src, dst, nrm, xw, ori, Ee);

    // conv2: h2 = agg(relu(h1) @ W2) + b2 + ori  -> out
    k_prep        <<<prep_blk, TB, 0, stream>>>(ori, Apk, Nn, n_tiles, 1);
    k_gemm<true>  <<<gemm_blk, TB, 0, stream>>>(Apk, Wb2, deg, b2, ori, 1.0f,
                                                xw, out, Nn);
    k_scatter     <<<scat_blk, TB, 0, stream>>>(src, dst, nrm, xw, out, Ee);

    // conv3: h3 = agg(relu(h2) @ W2) + b2 + ori  -> out
    k_prep        <<<prep_blk, TB, 0, stream>>>(out, Apk, Nn, n_tiles, 1);
    k_gemm<true>  <<<gemm_blk, TB, 0, stream>>>(Apk, Wb2, deg, b2, ori, 1.0f,
                                                xw, out, Nn);
    k_scatter     <<<scat_blk, TB, 0, stream>>>(src, dst, nrm, xw, out, Ee);
}
